// Lemmatizer_66554813219417
// MI455X (gfx1250) — compile-verified, matrix operand order checked by reading, not run on hardware
//
// CDNA5 / gfx1250 implementation of the lemmatizer forward pass.
// All GEMM-shaped work goes through v_wmma_f32_16x16x32_f16 (f16 in, f32 acc).
// Round-1 improvements:
//  - masked A rows read a zero page via pointer select (no exec-mask branch /
//    zero-fill v_movs in the inner loop)
//  - each wave computes two adjacent 16x16 N-tiles: A fragment reused, two
//    independent WMMA accumulator chains (fewer hazard NOPs, fewer loads/flop)
// d_in unpack assumes jax tree (sorted dict key) flattening order.

#include <hip/hip_runtime.h>
#include <math.h>

typedef __attribute__((ext_vector_type(16))) _Float16 v16h;
typedef __attribute__((ext_vector_type(8)))  _Float16 v8h;
typedef __attribute__((ext_vector_type(8)))  float    v8f;

#define DEVFN __device__ __forceinline__

static constexpr int kB   = 2048;
static constexpr int kS   = 24;
static constexpr int kT   = 20;
static constexpr int kV   = 128;
static constexpr int kBS  = kB * kS;     // 49152
static constexpr int BIGP = 1 << 30;     // "no periodic masking"

union AFrag { v16h v; v8h h[2]; };

DEVFN v16h frag_load(const _Float16* p) {
  // 16 halfs of one WMMA operand: k..k+7 and k+16..k+23 (hi lanes offset +8).
  AFrag f;
  f.h[0] = *(const v8h*)p;
  f.h[1] = *(const v8h*)(p + 16);
  return f.v;
}

// C[M,N] (f32, ldc) = (acc ? C : 0) + A[M,K]f16 * W[N,K]f16^T (+ bias)
// A row r reads physical row r+shift, valid iff 0 <= (r%period)+shift < period
// (conv1d taps; period=BIGP,shift=0 otherwise). Invalid rows read the zero
// page Z (>= K+32 halfs of zeros). K multiple of 32, N multiple of 16.
// Wave computes a 16x32 C tile (two N-tiles sharing the A fragment).
__global__ void __launch_bounds__(256)
gemm_f16(const _Float16* __restrict__ A, long lda,
         const _Float16* __restrict__ W, long ldw,
         float* __restrict__ C, long ldc,
         const float* __restrict__ bias,
         const _Float16* __restrict__ Z,
         int M, int N, int K, int acc, int shift, int period, int Nvalid) {
  const int wave = threadIdx.x >> 5;   // 8 waves / block (wave32)
  const int lane = threadIdx.x & 31;
  const int l    = lane & 15;
  const int hi   = lane >> 4;
  const int mbase = blockIdx.y * 32 + (wave & 1) * 16;
  const int nb0   = blockIdx.x * 128 + (wave >> 1) * 32;
  if (mbase >= M || nb0 >= N) return;
  const bool t1  = (nb0 + 16) < N;
  const int  nb1 = t1 ? (nb0 + 16) : nb0;

  const int arow = mbase + l;
  const int sp = arow % period + shift;
  const bool avalid = (sp >= 0) && (sp < period);
  const _Float16* pa = (avalid ? (A + (long)(arow + shift) * lda) : Z) + (hi ? 8 : 0);
  const _Float16* pw0 = W + (long)(nb0 + l) * ldw + (hi ? 8 : 0);
  const _Float16* pw1 = W + (long)(nb1 + l) * ldw + (hi ? 8 : 0);

  v8f c0 = {}, c1 = {};
  const long crow = mbase + hi * 8;
  const int  cc0 = nb0 + l;
  const int  cc1 = nb1 + l;
  if (acc) {
#pragma unroll
    for (int i = 0; i < 8; ++i) c0[i] = C[(crow + i) * ldc + cc0];
    if (t1) {
#pragma unroll
      for (int i = 0; i < 8; ++i) c1[i] = C[(crow + i) * ldc + cc1];
    }
  }
  for (int k = 0; k < K; k += 32) {
    v16h a  = frag_load(pa  + k);
    v16h b0 = frag_load(pw0 + k);
    v16h b1 = frag_load(pw1 + k);
    c0 = __builtin_amdgcn_wmma_f32_16x16x32_f16(false, a, false, b0,
                                                (short)0, c0, false, false);
    c1 = __builtin_amdgcn_wmma_f32_16x16x32_f16(false, a, false, b1,
                                                (short)0, c1, false, false);
  }
  if (cc0 < Nvalid) {
    const float bv = bias ? bias[cc0] : 0.f;
#pragma unroll
    for (int i = 0; i < 8; ++i) C[(crow + i) * ldc + cc0] = c0[i] + bv;
  }
  if (t1 && cc1 < Nvalid) {
    const float bv = bias ? bias[cc1] : 0.f;
#pragma unroll
    for (int i = 0; i < 8; ++i) C[(crow + i) * ldc + cc1] = c1[i] + bv;
  }
}

__global__ void fill_zero_k(unsigned int* p, size_t n) {
  size_t i = (size_t)blockIdx.x * blockDim.x + threadIdx.x;
  size_t st = (size_t)gridDim.x * blockDim.x;
  for (; i < n; i += st) p[i] = 0u;
}

// f32 -> f16 convert with generic gather (handles transpose + conv tap slices)
__global__ void cvt_f16(const float* __restrict__ src, _Float16* __restrict__ dst,
                        int rows, int cols, long srs, long scs, long ds) {
  long idx = (long)blockIdx.x * blockDim.x + threadIdx.x;
  if (idx >= (long)rows * cols) return;
  int r = (int)(idx / cols), c = (int)(idx % cols);
  dst[(long)r * ds + c] = (_Float16)src[(long)r * srs + (long)c * scs];
}

// embeddings -> x0 f16 [BS,224] (212 valid) ; cond f16/f32 [BS,112]
__global__ void embed_k(const int* __restrict__ ch, const int* __restrict__ cs,
                        const int* __restrict__ up, const int* __restrict__ lg,
                        const float* __restrict__ Ec, const float* __restrict__ Ecs,
                        const float* __restrict__ Eup, const float* __restrict__ Elg,
                        _Float16* __restrict__ x0, _Float16* __restrict__ c16,
                        float* __restrict__ c32) {
  int r = blockIdx.x, j = threadIdx.x;
  if (j >= 224) return;
  float v;
  if (j < 100)      v = Ec [(long)ch[r] * 100 + j];
  else if (j < 116) v = Ecs[(long)cs[r] * 16  + (j - 100)];
  else if (j < 180) v = Eup[(long)up[r] * 64  + (j - 116)];
  else if (j < 212) v = Elg[(long)lg[r] * 32  + (j - 180)];
  else              v = 0.f;
  x0[(long)r * 224 + j] = (_Float16)v;
  if (j >= 100 && j < 212) {
    int jc = j - 100;
    c16[(long)r * 112 + jc] = (_Float16)v;
    c32[(long)r * 112 + jc] = v;
  }
}

// conv bias + BN(eval) + relu, f32 in -> f16 out (stride 512)
__global__ void bn_relu_k(const float* __restrict__ x, const float* __restrict__ b,
                          const float* __restrict__ g, const float* __restrict__ be,
                          const float* __restrict__ m, const float* __restrict__ vv,
                          _Float16* __restrict__ y, long total, int C) {
  long idx = (long)blockIdx.x * blockDim.x + threadIdx.x;
  if (idx >= total) return;
  int ch = (int)(idx % C);
  float t = x[idx] + b[ch];
  t = g[ch] * (t - m[ch]) * rsqrtf(vv[ch] + 1e-5f) + be[ch];
  y[idx] = (_Float16)fmaxf(t, 0.f);
}

DEVFN float sigm(float x) { return 1.f / (1.f + __expf(-x)); }

// LSTM gates (PyTorch i,f,g,o order): c,h update; writes h as f16 (+opt f32)
__global__ void lstm_gate_k(const float* __restrict__ g, long gstride,
                            float* __restrict__ c,
                            _Float16* __restrict__ h16, long h16s,
                            float* __restrict__ h32, long h32s,
                            int Btot, int H) {
  long idx = (long)blockIdx.x * blockDim.x + threadIdx.x;
  if (idx >= (long)Btot * H) return;
  int b = (int)(idx / H), j = (int)(idx % H);
  const float* gr = g + (long)b * gstride;
  float gi = gr[j], gf = gr[H + j], gg = gr[2 * H + j], go = gr[3 * H + j];
  float cn = sigm(gf) * c[idx] + sigm(gi) * tanhf(gg);
  float hn = sigm(go) * tanhf(cn);
  c[idx] = cn;
  h16[(long)b * h16s + j] = (_Float16)hn;
  if (h32) h32[(long)b * h32s + j] = hn;
}

// [hf(256)|hb(256)|cond(112)|pad] -> 640 f16 (+ optional 624 f32 enc_out)
__global__ void concat_enc_k(const _Float16* __restrict__ hf, const _Float16* __restrict__ hb,
                             const _Float16* __restrict__ c16,
                             const float* __restrict__ hf32, const float* __restrict__ hb32,
                             const float* __restrict__ c32,
                             _Float16* __restrict__ o16, float* __restrict__ o32) {
  int r = blockIdx.x;
  for (int j = threadIdx.x; j < 640; j += blockDim.x) {
    _Float16 v;
    if (j < 256)      v = hf[(long)r * 256 + j];
    else if (j < 512) v = hb[(long)r * 256 + (j - 256)];
    else if (j < 624) v = c16[(long)r * 112 + (j - 512)];
    else              v = (_Float16)0.f;
    o16[(long)r * 640 + j] = v;
    if (o32 && j < 624) {
      float f;
      if (j < 256)      f = hf32[(long)r * 256 + j];
      else if (j < 512) f = hb32[(long)r * 256 + (j - 256)];
      else              f = c32 [(long)r * 112 + (j - 512)];
      o32[(long)r * 624 + j] = f;
    }
  }
}

// per-batch attention + decoder input assembly: xin = [f16(ctx 624)|prev_emb|pad]
__global__ void attention_k(const float* __restrict__ encp, const float* __restrict__ dWd,
                            const float* __restrict__ av, const float* __restrict__ eo,
                            const float* __restrict__ Echar, const int* __restrict__ oidx,
                            _Float16* __restrict__ xin, int tstep) {
  int b = blockIdx.x, tid = threadIdx.x;
  __shared__ float e[24];
  __shared__ float att[24];
  if (tid < 24) {
    float acc = 0.f;
    const float* ep = encp + ((long)b * 24 + tid) * 112;
    const float* dp = dWd + (long)b * 112;
    for (int k = 0; k < 100; ++k) acc += av[k] * tanhf(ep[k] + dp[k]);
    e[tid] = acc;
  }
  __syncthreads();
  if (tid == 0) {
    float mx = e[0];
    for (int s = 1; s < 24; ++s) mx = fmaxf(mx, e[s]);
    float sum = 0.f;
    for (int s = 0; s < 24; ++s) { att[s] = __expf(e[s] - mx); sum += att[s]; }
    float inv = 1.f / sum;
    for (int s = 0; s < 24; ++s) att[s] *= inv;
  }
  __syncthreads();
  for (int d = tid; d < 624; d += blockDim.x) {
    float acc = 0.f;
    const float* p = eo + (long)b * 24 * 624 + d;
    for (int s = 0; s < 24; ++s) acc += att[s] * p[s * 624];
    xin[(long)b * 736 + d] = (_Float16)acc;
  }
  int pc = (tstep > 0) ? oidx[(long)b * kT + (tstep - 1)] : 0;
  for (int j = tid; j < 112; j += blockDim.x) {
    float v = (tstep > 0 && j < 100) ? Echar[(long)pc * 100 + j] : 0.f;
    xin[(long)b * 736 + 624 + j] = (_Float16)v;
  }
}

__global__ void fill_start_k(const float* __restrict__ s, _Float16* __restrict__ xin) {
  long idx = (long)blockIdx.x * blockDim.x + threadIdx.x;
  if (idx >= (long)kB * 736) return;
  int j = (int)(idx % 736);
  xin[idx] = (_Float16)((j < 724) ? s[j] : 0.f);
}

// --- d_in indices under jax tree (sorted-key) flattening ------------------
enum {
  I_CASE = 0, I_CHAR = 1, I_LANG = 2, I_OIDX = 3,
  I_ATT_WD = 4, I_ATT_WE = 5, I_ATT_V = 6,
  I_CONV0 = 7, I_CONV1 = 13, I_CONV2 = 19,          // each: b,be,g,m,v,w
  I_DEC0_B = 25, I_DEC0_WHH = 26, I_DEC0_WIH = 27,
  I_DEC1_B = 28, I_DEC1_WHH = 29, I_DEC1_WIH = 30,
  I_EMB_CASE = 31, I_EMB_CHAR = 32, I_EMB_LANG = 33, I_EMB_UPOS = 34,
  I_E0_BB = 35, I_E0_BF = 36, I_E0_HHB = 37, I_E0_HHF = 38, I_E0_IHB = 39, I_E0_IHF = 40,
  I_E1_BB = 41, I_E1_BF = 42, I_E1_HHB = 43, I_E1_HHF = 44, I_E1_IHB = 45, I_E1_IHF = 46,
  I_OCS_B = 47, I_OCS_W = 48, I_OC_B = 49, I_OC_W = 50, I_START = 51, I_UPOS = 52
};

extern "C" void kernel_launch(void* const* d_in, const int* in_sizes, int n_in,
                              void* d_out, int out_size, void* d_ws, size_t ws_size,
                              hipStream_t stream) {
  (void)in_sizes; (void)n_in; (void)out_size;
  const int* case_idx = (const int*)d_in[I_CASE];
  const int* char_idx = (const int*)d_in[I_CHAR];
  const int* lang_idx = (const int*)d_in[I_LANG];
  const int* oidx     = (const int*)d_in[I_OIDX];
  const int* upos_idx = (const int*)d_in[I_UPOS];
  const float* f_attWd = (const float*)d_in[I_ATT_WD];
  const float* f_attWe = (const float*)d_in[I_ATT_WE];
  const float* f_attv  = (const float*)d_in[I_ATT_V];
  const float* cb[3], *cbe[3], *cg[3], *cm[3], *cv[3], *cw[3];
  const int cbase[3] = { I_CONV0, I_CONV1, I_CONV2 };
  for (int i = 0; i < 3; ++i) {
    cb[i]  = (const float*)d_in[cbase[i] + 0];
    cbe[i] = (const float*)d_in[cbase[i] + 1];
    cg[i]  = (const float*)d_in[cbase[i] + 2];
    cm[i]  = (const float*)d_in[cbase[i] + 3];
    cv[i]  = (const float*)d_in[cbase[i] + 4];
    cw[i]  = (const float*)d_in[cbase[i] + 5];
  }

  char* base = (char*)d_ws;
  size_t off = 0;
  auto alloc = [&](size_t bytes) -> char* {
    off = (off + 255) & ~(size_t)255;
    char* p = base + off;
    off += bytes;
    return p;
  };
  auto af16 = [&](size_t elems) { return (_Float16*)alloc(elems * 2); };
  auto af32 = [&](size_t elems) { return (float*)alloc(elems * 4); };

  // f16 weight arena (pre-zeroed => K/N padding stays zero)
  _Float16* wconv[3];
  wconv[0] = af16((size_t)5 * 512 * 224);
  wconv[1] = af16((size_t)5 * 512 * 512);
  wconv[2] = af16((size_t)5 * 512 * 512);
  _Float16* We0ihf = af16((size_t)1024 * 512);
  _Float16* We0hhf = af16((size_t)1024 * 256);
  _Float16* We0ihb = af16((size_t)1024 * 512);
  _Float16* We0hhb = af16((size_t)1024 * 256);
  _Float16* We1ihf = af16((size_t)1024 * 640);
  _Float16* We1hhf = af16((size_t)1024 * 256);
  _Float16* We1ihb = af16((size_t)1024 * 640);
  _Float16* We1hhb = af16((size_t)1024 * 256);
  _Float16* Wd0ih  = af16((size_t)2048 * 736);
  _Float16* Wd0hh  = af16((size_t)2048 * 512);
  _Float16* Wd1ih  = af16((size_t)2048 * 512);
  _Float16* Wd1hh  = af16((size_t)2048 * 512);
  _Float16* WeT    = af16((size_t)112 * 640);   // att_We^T, rows 100..111 zero
  _Float16* WdT    = af16((size_t)112 * 512);
  _Float16* WocT   = af16((size_t)128 * 512);   // out_char_w^T
  _Float16* WocsT  = af16((size_t)16 * 512);    // out_case_w^T, rows 3..15 zero
  _Float16* zeroP  = af16(1024);                // zero page (masked rows / zero h)

  // activations
  _Float16* x0     = af16((size_t)kBS * 224);
  _Float16* cond16 = af16((size_t)kBS * 112);
  float*    cond32 = af32((size_t)kBS * 112);
  float*    convA  = af32((size_t)kBS * 512);
  _Float16* convH1 = af16((size_t)kBS * 512);
  _Float16* convH2 = af16((size_t)kBS * 512);
  float*    gbuf   = af32((size_t)kB * 2048);
  _Float16* hf16   = af16((size_t)kBS * 256);
  _Float16* hb16   = af16((size_t)kBS * 256);
  float*    hf32   = af32((size_t)kBS * 256);
  float*    hb32   = af32((size_t)kBS * 256);
  _Float16* encin2 = af16((size_t)kBS * 640);
  _Float16* eo16   = af16((size_t)kBS * 640);
  float*    eo32   = af32((size_t)kBS * 624);
  float*    encp   = af32((size_t)kBS * 112);
  float*    c_enc  = af32((size_t)kB * 256);
  float*    c1d    = af32((size_t)kB * 512);
  float*    c2d    = af32((size_t)kB * 512);
  _Float16* h1f    = af16((size_t)kB * 512);
  _Float16* h2f    = af16((size_t)kB * 512);
  _Float16* xin    = af16((size_t)kB * 736);
  float*    dWd    = af32((size_t)kB * 112);

  // 1) zero the whole workspace (covers weight pads, zero page, decoder c init)
  fill_zero_k<<<4096, 256, 0, stream>>>((unsigned int*)d_ws, ws_size >> 2);

  auto zero = [&](void* p, size_t elems32) {
    fill_zero_k<<<1024, 256, 0, stream>>>((unsigned int*)p, elems32);
  };
  auto cvt = [&](const float* s, _Float16* d, int r, int c, long srs, long scs, long ds) {
    long n = (long)r * c;
    cvt_f16<<<(unsigned)((n + 255) / 256), 256, 0, stream>>>(s, d, r, c, srs, scs, ds);
  };
  auto gemm = [&](const _Float16* A, long lda, const _Float16* W, long ldw,
                  float* C, long ldc, const float* bias,
                  int M, int N, int K, int acc, int shift, int period, int Nvalid) {
    dim3 grid((N + 127) / 128, (M + 31) / 32);
    gemm_f16<<<grid, 256, 0, stream>>>(A, lda, W, ldw, C, ldc, bias, zeroP,
                                       M, N, K, acc, shift, period, Nvalid);
  };
  auto gate = [&](float* g, long gs, float* c, _Float16* h16p, long h16s,
                  float* h32p, long h32s, int Bt, int H) {
    long n = (long)Bt * H;
    lstm_gate_k<<<(unsigned)((n + 255) / 256), 256, 0, stream>>>(
        g, gs, c, h16p, h16s, h32p, h32s, Bt, H);
  };

  // 2) weight conversion (f32 -> f16, K-padded, transposed where needed)
  for (int ci = 0; ci < 3; ++ci) {
    int cin = (ci == 0) ? 212 : 512, cinp = (ci == 0) ? 224 : 512;
    for (int t = 0; t < 5; ++t)
      cvt(cw[ci] + t, wconv[ci] + (long)t * 512 * cinp, 512, cin, (long)cin * 5, 5, cinp);
  }
  cvt((const float*)d_in[I_E0_IHF], We0ihf, 1024, 512, 512, 1, 512);
  cvt((const float*)d_in[I_E0_HHF], We0hhf, 1024, 256, 256, 1, 256);
  cvt((const float*)d_in[I_E0_IHB], We0ihb, 1024, 512, 512, 1, 512);
  cvt((const float*)d_in[I_E0_HHB], We0hhb, 1024, 256, 256, 1, 256);
  cvt((const float*)d_in[I_E1_IHF], We1ihf, 1024, 624, 624, 1, 640);
  cvt((const float*)d_in[I_E1_HHF], We1hhf, 1024, 256, 256, 1, 256);
  cvt((const float*)d_in[I_E1_IHB], We1ihb, 1024, 624, 624, 1, 640);
  cvt((const float*)d_in[I_E1_HHB], We1hhb, 1024, 256, 256, 1, 256);
  cvt((const float*)d_in[I_DEC0_WIH], Wd0ih, 2048, 724, 724, 1, 736);
  cvt((const float*)d_in[I_DEC0_WHH], Wd0hh, 2048, 512, 512, 1, 512);
  cvt((const float*)d_in[I_DEC1_WIH], Wd1ih, 2048, 512, 512, 1, 512);
  cvt((const float*)d_in[I_DEC1_WHH], Wd1hh, 2048, 512, 512, 1, 512);
  cvt(f_attWe, WeT, 100, 624, 1, 100, 640);                       // transpose
  cvt(f_attWd, WdT, 100, 512, 1, 100, 512);                       // transpose
  cvt((const float*)d_in[I_OC_W],  WocT,  128, 512, 1, 128, 512); // transpose
  cvt((const float*)d_in[I_OCS_W], WocsT, 3,   512, 1, 3,   512); // transpose

  // 3) embeddings
  embed_k<<<kBS, 256, 0, stream>>>(char_idx, case_idx, upos_idx, lang_idx,
                                   (const float*)d_in[I_EMB_CHAR],
                                   (const float*)d_in[I_EMB_CASE],
                                   (const float*)d_in[I_EMB_UPOS],
                                   (const float*)d_in[I_EMB_LANG],
                                   x0, cond16, cond32);

  // 4) convs: 5 shifted accumulating WMMA GEMMs each, then bias+BN+relu
  auto conv = [&](const _Float16* X, int Kp, _Float16* W, int ci, _Float16* Y) {
    for (int t = 0; t < 5; ++t)
      gemm(X, Kp, W + (long)t * 512 * Kp, Kp, convA, 512, nullptr,
           kBS, 512, Kp, t > 0, t - 2, kS, 512);
    long tot = (long)kBS * 512;
    bn_relu_k<<<(unsigned)((tot + 255) / 256), 256, 0, stream>>>(
        convA, cb[ci], cg[ci], cbe[ci], cm[ci], cv[ci], Y, tot, 512);
  };
  conv(x0, 224, wconv[0], 0, convH1);
  conv(convH1, 512, wconv[1], 1, convH2);
  conv(convH2, 512, wconv[2], 2, convH1);

  // 5) BiLSTM encoder (per-step GEMMs; recurrent term accumulates via WMMA C)
  auto enc_dir = [&](const _Float16* X, int Kp, const _Float16* Wih,
                     const float* bias, const _Float16* Whh,
                     _Float16* hseq, float* h32seq, bool rev) {
    zero(c_enc, (size_t)kB * 256);
    for (int step = 0; step < kS; ++step) {
      int s = rev ? (kS - 1 - step) : step;
      gemm(X + (long)s * Kp, (long)kS * Kp, Wih, Kp, gbuf, 1024, bias,
           kB, 1024, Kp, 0, 0, BIGP, 1024);
      if (step == 0)
        gemm(zeroP, 0, Whh, 256, gbuf, 1024, nullptr, kB, 1024, 256, 1, 0, BIGP, 1024);
      else {
        int sp = rev ? (s + 1) : (s - 1);
        gemm(hseq + (long)sp * 256, (long)kS * 256, Whh, 256, gbuf, 1024, nullptr,
             kB, 1024, 256, 1, 0, BIGP, 1024);
      }
      gate(gbuf, 1024, c_enc, hseq + (long)s * 256, (long)kS * 256,
           h32seq ? h32seq + (long)s * 256 : nullptr, (long)kS * 256, kB, 256);
    }
  };
  // layer 1 (input: conv out, K=512)
  enc_dir(convH1, 512, We0ihf, (const float*)d_in[I_E0_BF], We0hhf, hf16, nullptr, false);
  enc_dir(convH1, 512, We0ihb, (const float*)d_in[I_E0_BB], We0hhb, hb16, nullptr, true);
  concat_enc_k<<<kBS, 256, 0, stream>>>(hf16, hb16, cond16, nullptr, nullptr, nullptr,
                                        encin2, nullptr);
  // layer 2 (input: 640-padded concat; also keep f32 h for attention context)
  enc_dir(encin2, 640, We1ihf, (const float*)d_in[I_E1_BF], We1hhf, hf16, hf32, false);
  enc_dir(encin2, 640, We1ihb, (const float*)d_in[I_E1_BB], We1hhb, hb16, hb32, true);
  concat_enc_k<<<kBS, 256, 0, stream>>>(hf16, hb16, cond16, hf32, hb32, cond32,
                                        eo16, eo32);

  // 6) attention key precompute: enc_proj = enc_out * att_We  [BS,112]
  gemm(eo16, 640, WeT, 640, encp, 112, nullptr, kBS, 112, 640, 0, 0, BIGP, 112);

  // 7) decoder (c1d/c2d already zero from workspace clear)
  auto dec_step = [&](const _Float16* h1p, long h1lda, const _Float16* h2p, long h2lda) {
    gemm(xin, 736, Wd0ih, 736, gbuf, 2048, (const float*)d_in[I_DEC0_B],
         kB, 2048, 736, 0, 0, BIGP, 2048);
    gemm(h1p, h1lda, Wd0hh, 512, gbuf, 2048, nullptr, kB, 2048, 512, 1, 0, BIGP, 2048);
    gate(gbuf, 2048, c1d, h1f, 512, nullptr, 0, kB, 512);
    gemm(h1f, 512, Wd1ih, 512, gbuf, 2048, (const float*)d_in[I_DEC1_B],
         kB, 2048, 512, 0, 0, BIGP, 2048);
    gemm(h2p, h2lda, Wd1hh, 512, gbuf, 2048, nullptr, kB, 2048, 512, 1, 0, BIGP, 2048);
    gate(gbuf, 2048, c2d, h2f, 512, nullptr, 0, kB, 512);
  };
  // prime with start frame, zero hidden state
  {
    long n = (long)kB * 736;
    fill_start_k<<<(unsigned)((n + 255) / 256), 256, 0, stream>>>(
        (const float*)d_in[I_START], xin);
  }
  dec_step(zeroP, 0, zeroP, 0);

  float* outC = (float*)d_out;                 // [B,T,128]
  float* outS = outC + (long)kB * kT * kV;     // [B,T,3]
  for (int t = 0; t < kT; ++t) {
    gemm(h2f, 512, WdT, 512, dWd, 112, nullptr, kB, 112, 512, 0, 0, BIGP, 112);
    attention_k<<<kB, 256, 0, stream>>>(encp, dWd, f_attv, eo32,
                                        (const float*)d_in[I_EMB_CHAR], oidx, xin, t);
    dec_step(h1f, 512, h2f, 512);
    gemm(h2f, 512, WocT, 512, outC + (long)t * kV, (long)kT * kV,
         (const float*)d_in[I_OC_B], kB, 128, 512, 0, 0, BIGP, 128);
    gemm(h2f, 512, WocsT, 512, outS + (long)t * 3, (long)kT * 3,
         (const float*)d_in[I_OCS_B], kB, 16, 512, 0, 0, BIGP, 3);
  }
}
